// RecurrentBaseline_5669356833833
// MI455X (gfx1250) — matrix operand order repaired and checked by reading, not compiled, not run on hardware
//
#include <hip/hip_runtime.h>
#include <hip/hip_bf16.h>

typedef __bf16 bf16;
typedef __attribute__((ext_vector_type(16))) __bf16 v16bf;
typedef __attribute__((ext_vector_type(8)))  __bf16 v8bf;
typedef __attribute__((ext_vector_type(8)))  float  v8f;

#define B_    256
#define A_    5
#define T_    64
#define NIN_  4
#define NHID_ 256
#define L_    2
#define H_    1280      // A_*NHID_
#define G4H_  5120      // 4*H_

constexpr int MODE_F32       = 0;
constexpr int MODE_RELU_BF16 = 1;

// ---------------- elementwise helpers ----------------

__global__ void k_convert_bf16(const float* __restrict__ src, bf16* __restrict__ dst, int n) {
  for (int i = blockIdx.x * blockDim.x + threadIdx.x; i < n; i += gridDim.x * blockDim.x)
    dst[i] = (bf16)src[i];
}

__global__ void k_zero_state(bf16* __restrict__ h, float* __restrict__ c, int n) {
  int i = blockIdx.x * blockDim.x + threadIdx.x;
  if (i < n) { h[i] = (bf16)0.0f; c[i] = 0.0f; }
}

// ins = (t % ps == 0) ? inputs[:,:,t,:] : prev_out   (teacher forcing)
__global__ void k_compute_ins(const float* __restrict__ inputs, const float* __restrict__ prev_out,
                              float* __restrict__ ins, int t, const int* __restrict__ psPtr) {
  int idx = blockIdx.x * blockDim.x + threadIdx.x;
  if (idx >= B_ * A_ * NIN_) return;
  int ps = psPtr[0];
  int i = idx % NIN_;
  int a = (idx / NIN_) % A_;
  int b = idx / (NIN_ * A_);
  float v;
  if ((t % ps) == 0) v = inputs[((size_t)(b * A_ + a) * T_ + t) * NIN_ + i];
  else               v = prev_out[idx];
  ins[idx] = v;
}

// x1 = relu(ins @ w1_1^T + b1_1)  (K=4, trivial VALU kernel, stores bf16)
__global__ void k_layer1_1(const float* __restrict__ ins, const float* __restrict__ w1,
                           const float* __restrict__ b1, bf16* __restrict__ x1) {
  int idx = blockIdx.x * blockDim.x + threadIdx.x;
  if (idx >= (B_ * A_) * NHID_) return;
  int j   = idx % NHID_;
  int row = idx / NHID_;
  const float* iv = ins + row * NIN_;
  const float* wv = w1 + j * NIN_;
  float acc = b1[j];
#pragma unroll
  for (int k = 0; k < NIN_; ++k) acc += iv[k] * wv[k];
  x1[idx] = (bf16)fmaxf(acc, 0.0f);
}

// ---------------- block-tiled WMMA GEMM with async-LDS B staging ----------------
// C[M,N] = A[M,K] @ W[N,K]^T (+ A2 @ W2^T) + bias1 + bias2, optional relu.
// Block = 128 threads (4 waves). Block tile = 64(M) x 64(N); wave w owns M rows
// [m0+16w, m0+16w+16) across all 4 N-tiles (register-blocked: one A fragment
// feeds 4 WMMAs). The 64x32 B panel (shared by all 4 waves) is staged into LDS
// with global_load_async_to_lds_b128 (ASYNCcnt), double-buffered so staging of
// chunk i+1 overlaps the WMMAs of chunk i. DUAL is a compile-time flag so no
// EXEC-divergent branch ever surrounds a WMMA.
// Fragment layouts used here (wave32; A and B share one k order):
//   A: lane l -> row (l&15); 16 consecutive K starting at (l>>4)*16.
//   B: lane l -> col (l&15); 16 consecutive K starting at (l>>4)*16.
//   C/D: lane l -> col (l&15); vgpr r -> row r + 8*(l>>4).
template <int MODE, bool DUAL>
__global__ __launch_bounds__(128)
void k_wmma_gemm_blk(const bf16* __restrict__ A, const bf16* __restrict__ W,
                     const bf16* __restrict__ A2, const bf16* __restrict__ W2,
                     const float* __restrict__ bias1, const float* __restrict__ bias2,
                     void* __restrict__ outp, int M, int N, int K) {
  constexpr int NMAT = DUAL ? 2 : 1;
  __shared__ bf16 smem[2][NMAT][64 * 32];   // [double buf][matrix][64 N-rows x 32 K] = 4KB each

  const int tid  = threadIdx.x;             // 0..127
  const int wave = tid >> 5;
  const int lane = tid & 31;
  const int half = lane >> 4;
  const int lidx = lane & 15;

  const int n0 = blockIdx.x * 64;
  const int m0 = blockIdx.y * 64 + wave * 16;

  const bf16* arow  = A + (size_t)(m0 + lidx) * K;
  const bf16* a2row = DUAL ? (A2 + (size_t)(m0 + lidx) * K) : nullptr;

  // Stage one 64x32 (bf16) panel of W (and W2 if DUAL) into LDS buffer `buf`.
  // 128 threads x 2 (x NMAT) async b128 transfers = 4KB per matrix.
  auto stage = [&](int k0, int buf) {
    const int r0 = tid >> 2;   // 0..31
    const int q  = tid & 3;    // 16B quarter of the 64B row
#pragma unroll
    for (int s = 0; s < 2; ++s) {
      const int row = r0 + s * 32;   // 0..63
      {
        const bf16* gp = W + (size_t)(n0 + row) * K + k0 + q * 8;
        unsigned lp = (unsigned)(uintptr_t)&smem[buf][0][row * 32 + q * 8];
        asm volatile("global_load_async_to_lds_b128 %0, %1, off"
                     :: "v"(lp), "v"((unsigned long long)(uintptr_t)gp) : "memory");
      }
      if (DUAL) {
        const bf16* gp2 = W2 + (size_t)(n0 + row) * K + k0 + q * 8;
        unsigned lp2 = (unsigned)(uintptr_t)&smem[buf][1][row * 32 + q * 8];
        asm volatile("global_load_async_to_lds_b128 %0, %1, off"
                     :: "v"(lp2), "v"((unsigned long long)(uintptr_t)gp2) : "memory");
      }
    }
  };

  v8f acc[4] = {v8f{}, v8f{}, v8f{}, v8f{}};

  const int nk = K / 32;
  stage(0, 0);
  for (int i = 0; i < nk; ++i) {
    asm volatile("s_wait_asynccnt 0x0" ::: "memory");
    __syncthreads();                               // panel i visible to all waves
    if (i + 1 < nk) stage((i + 1) * 32, (i + 1) & 1);  // overlaps WMMAs below
    const int k0  = i * 32;
    const int buf = i & 1;

    v8bf alo = *(const v8bf*)(arow + k0 + half * 16);
    v8bf ahi = *(const v8bf*)(arow + k0 + half * 16 + 8);
    v16bf av = __builtin_shufflevector(alo, ahi, 0,1,2,3,4,5,6,7,8,9,10,11,12,13,14,15);
#pragma unroll
    for (int j = 0; j < 4; ++j) {
      v16bf bv = *(const v16bf*)&smem[buf][0][(j * 16 + lidx) * 32 + half * 16];
      acc[j] = __builtin_amdgcn_wmma_f32_16x16x32_bf16(false, av, false, bv, (short)0,
                                                       acc[j], false, false);
    }
    if (DUAL) {
      v8bf alo2 = *(const v8bf*)(a2row + k0 + half * 16);
      v8bf ahi2 = *(const v8bf*)(a2row + k0 + half * 16 + 8);
      v16bf av2 = __builtin_shufflevector(alo2, ahi2, 0,1,2,3,4,5,6,7,8,9,10,11,12,13,14,15);
#pragma unroll
      for (int j = 0; j < 4; ++j) {
        v16bf bv2 = *(const v16bf*)&smem[buf][NMAT - 1][(j * 16 + lidx) * 32 + half * 16];
        acc[j] = __builtin_amdgcn_wmma_f32_16x16x32_bf16(false, av2, false, bv2, (short)0,
                                                         acc[j], false, false);
      }
    }
    __syncthreads();                               // done reading buf before it is re-staged
  }

#pragma unroll
  for (int j = 0; j < 4; ++j) {
    const int n = n0 + j * 16 + lidx;
    float badd = 0.0f;
    if (bias1) badd += bias1[n];
    if (bias2) badd += bias2[n];
#pragma unroll
    for (int r = 0; r < 8; ++r) {
      const int m = m0 + r + 8 * half;
      float v = acc[j][r] + badd;
      if (MODE == MODE_RELU_BF16) ((bf16*)outp)[(size_t)m * N + n] = (bf16)fmaxf(v, 0.0f);
      else                        ((float*)outp)[(size_t)m * N + n] = v;
    }
  }
}

// Small single-wave-per-tile GEMM for the N=20 head (guarded columns, no dual path).
__global__ void k_wmma_gemm_small(const bf16* __restrict__ A, const bf16* __restrict__ W,
                                  const float* __restrict__ bias, float* __restrict__ outp,
                                  int M, int N, int K) {
  const int tileN = blockIdx.x * 16;
  const int tileM = blockIdx.y * 16;
  const int lane  = threadIdx.x & 31;
  const int half  = lane >> 4;
  const int lidx  = lane & 15;

  const bf16* arow = A + (size_t)(tileM + lidx) * K;
  int nrow = tileN + lidx;
  if (nrow > N - 1) nrow = N - 1;                // clamp; store guarded below
  const bf16* wrow = W + (size_t)nrow * K;

  v8f acc = {};
  for (int k0 = 0; k0 < K; k0 += 32) {
    v8bf alo = *(const v8bf*)(arow + k0 + half * 16);
    v8bf ahi = *(const v8bf*)(arow + k0 + half * 16 + 8);
    v16bf av = __builtin_shufflevector(alo, ahi, 0,1,2,3,4,5,6,7,8,9,10,11,12,13,14,15);
    v16bf bv = *(const v16bf*)(wrow + k0 + half * 16);
    acc = __builtin_amdgcn_wmma_f32_16x16x32_bf16(false, av, false, bv, (short)0, acc, false, false);
  }
  const int n = tileN + lidx;
  if (n >= N) return;
  float badd = bias ? bias[n] : 0.0f;
#pragma unroll
  for (int r = 0; r < 8; ++r) {
    const int m = tileM + r + 8 * half;
    outp[(size_t)m * N + n] = acc[r] + badd;
  }
}

// ---------------- LSTM gating ----------------
__global__ void k_lstm_gate(const float* __restrict__ gates, float* __restrict__ c,
                            bf16* __restrict__ h) {
  int idx = blockIdx.x * blockDim.x + threadIdx.x;
  if (idx >= B_ * H_) return;
  int b = idx / H_, j = idx % H_;
  const float* g = gates + (size_t)b * G4H_;
  float ig = g[j], fg = g[H_ + j], gg = g[2 * H_ + j], og = g[3 * H_ + j];
  float si = 1.0f / (1.0f + __expf(-ig));
  float sf = 1.0f / (1.0f + __expf(-fg));
  float so = 1.0f / (1.0f + __expf(-og));
  float tg = tanhf(gg);
  float cn = sf * c[idx] + si * tg;
  c[idx] = cn;
  h[idx] = (bf16)(so * tanhf(cn));
}

// out = y3 + ins (residual); scatter to d_out[b][a][t][i]; keep prev_out.
__global__ void k_finalize(const float* __restrict__ y3, const float* __restrict__ ins,
                           float* __restrict__ out, float* __restrict__ prev_out, int t) {
  int idx = blockIdx.x * blockDim.x + threadIdx.x;
  if (idx >= B_ * A_ * NIN_) return;
  float v = y3[idx] + ins[idx];
  int i = idx % NIN_;
  int a = (idx / NIN_) % A_;
  int b = idx / (NIN_ * A_);
  out[((size_t)(b * A_ + a) * (T_ - 1) + t) * NIN_ + i] = v;
  prev_out[idx] = v;
}

// ---------------- host launcher ----------------

extern "C" void kernel_launch(void* const* d_in, const int* in_sizes, int n_in,
                              void* d_out, int out_size, void* d_ws, size_t ws_size,
                              hipStream_t stream) {
  (void)in_sizes; (void)n_in; (void)out_size; (void)ws_size;
  const float* inputs = (const float*)d_in[0];
  const float* w1_1   = (const float*)d_in[1];
  const float* b1_1   = (const float*)d_in[2];
  const float* w1_2   = (const float*)d_in[3];
  const float* b1_2   = (const float*)d_in[4];
  const float* w_ih   = (const float*)d_in[5];
  const float* w_hh   = (const float*)d_in[6];
  const float* b_ih   = (const float*)d_in[7];
  const float* b_hh   = (const float*)d_in[8];
  const float* w2_1   = (const float*)d_in[9];
  const float* b2_1   = (const float*)d_in[10];
  const float* w2_2   = (const float*)d_in[11];
  const float* b2_2   = (const float*)d_in[12];
  const int*   psPtr  = (const int*)d_in[13];
  float* out = (float*)d_out;

  char* p = (char*)d_ws;
  auto carve = [&](size_t bytes) -> void* {
    void* r = (void*)p;
    p += (bytes + 255) & ~(size_t)255;
    return r;
  };
  const size_t WIH = (size_t)L_ * G4H_ * H_;
  bf16*  wih_bf = (bf16*)carve(WIH * 2);
  bf16*  whh_bf = (bf16*)carve(WIH * 2);
  bf16*  w12_bf = (bf16*)carve((size_t)NHID_ * NHID_ * 2);
  bf16*  w21_bf = (bf16*)carve((size_t)H_ * H_ * 2);
  bf16*  w22_bf = (bf16*)carve((size_t)A_ * NIN_ * H_ * 2);
  bf16*  x1     = (bf16*)carve((size_t)B_ * A_ * NHID_ * 2);
  bf16*  x2     = (bf16*)carve((size_t)B_ * H_ * 2);
  float* gates  = (float*)carve((size_t)B_ * G4H_ * 4);
  bf16*  hbuf   = (bf16*)carve((size_t)L_ * B_ * H_ * 2);
  float* cbuf   = (float*)carve((size_t)L_ * B_ * H_ * 4);
  bf16*  y2     = (bf16*)carve((size_t)B_ * H_ * 2);
  float* y3     = (float*)carve((size_t)B_ * A_ * NIN_ * 4);
  float* ins    = (float*)carve((size_t)B_ * A_ * NIN_ * 4);
  float* prev   = (float*)carve((size_t)B_ * A_ * NIN_ * 4);

  // one-time (per launch) weight down-conversion to bf16; L2-resident thereafter
  k_convert_bf16<<<2048, 256, 0, stream>>>(w_ih, wih_bf, (int)WIH);
  k_convert_bf16<<<2048, 256, 0, stream>>>(w_hh, whh_bf, (int)WIH);
  k_convert_bf16<<<256, 256, 0, stream>>>(w1_2, w12_bf, NHID_ * NHID_);
  k_convert_bf16<<<2048, 256, 0, stream>>>(w2_1, w21_bf, H_ * H_);
  k_convert_bf16<<<128, 256, 0, stream>>>(w2_2, w22_bf, A_ * NIN_ * H_);
  k_zero_state<<<(L_ * B_ * H_ + 255) / 256, 256, 0, stream>>>(hbuf, cbuf, L_ * B_ * H_);

  for (int t = 0; t < T_ - 1; ++t) {
    k_compute_ins<<<(B_ * A_ * NIN_ + 255) / 256, 256, 0, stream>>>(inputs, prev, ins, t, psPtr);
    k_layer1_1<<<(B_ * A_ * NHID_ + 255) / 256, 256, 0, stream>>>(ins, w1_1, b1_1, x1);

    // x2 = relu(x1 @ w1_2^T + b1_2): M=1280, N=256, K=256
    k_wmma_gemm_blk<MODE_RELU_BF16, false><<<dim3(NHID_ / 64, (B_ * A_) / 64), 128, 0, stream>>>(
        x1, w12_bf, nullptr, nullptr, b1_2, nullptr, x2, B_ * A_, NHID_, NHID_);

    for (int l = 0; l < L_; ++l) {
      const bf16* xin = (l == 0) ? x2 : hbuf;  // layer1 input = new h[0]
      // gates = x@Wih^T + h_prev@Whh^T + b_ih + b_hh : M=256, N=5120, K=1280 (fused dual GEMM)
      k_wmma_gemm_blk<MODE_F32, true><<<dim3(G4H_ / 64, B_ / 64), 128, 0, stream>>>(
          xin, wih_bf + (size_t)l * G4H_ * H_,
          hbuf + (size_t)l * B_ * H_, whh_bf + (size_t)l * G4H_ * H_,
          b_ih + l * G4H_, b_hh + l * G4H_,
          gates, B_, G4H_, H_);
      k_lstm_gate<<<(B_ * H_ + 255) / 256, 256, 0, stream>>>(
          gates, cbuf + (size_t)l * B_ * H_, hbuf + (size_t)l * B_ * H_);
    }

    // y2 = relu(h[1] @ w2_1^T + b2_1): M=256, N=1280, K=1280
    k_wmma_gemm_blk<MODE_RELU_BF16, false><<<dim3(H_ / 64, B_ / 64), 128, 0, stream>>>(
        hbuf + (size_t)B_ * H_, w21_bf, nullptr, nullptr, b2_1, nullptr, y2, B_, H_, H_);
    // y3 = y2 @ w2_2^T + b2_2: M=256, N=20 (guarded), K=1280
    k_wmma_gemm_small<<<dim3((A_ * NIN_ + 15) / 16, B_ / 16), 32, 0, stream>>>(
        y2, w22_bf, b2_2, y3, B_, A_ * NIN_, H_);
    k_finalize<<<(B_ * A_ * NIN_ + 255) / 256, 256, 0, stream>>>(y3, ins, out, prev, t);
  }
}
